// HybridTokenMixer_17961553231895
// MI455X (gfx1250) — matrix operand order repaired and checked by reading, not compiled, not run on hardware
//
#include <hip/hip_runtime.h>
#include <hip/hip_bf16.h>

// ---------- shapes ----------
#define Bsz   64
#define Cc    128
#define CH    64
#define Hh    56
#define Ww    56
#define Nn    3136     // 56*56
#define NK    196      // 14*14
#define NKP   208      // 13*16 (k-token pad for 16-wide tiles)
#define NKP2  224      // 7*32  (k-token pad for 32-deep K steps)
#define HEADS 2
#define HD    32

typedef __attribute__((ext_vector_type(16))) _Float16 v16h;
typedef __attribute__((ext_vector_type(8)))  _Float16 v8h;
typedef __attribute__((ext_vector_type(8)))  float    v8f;
typedef __attribute__((ext_vector_type(4)))  float    f4v;

// ---------- workspace layout (bytes, 256-aligned) ----------
constexpr size_t OFF_Y  = 0;                                   // f32 [B][128][3136]
constexpr size_t SZ_Y   = (size_t)Bsz*Cc*Nn*4;                 // 102,760,448
constexpr size_t OFF_Q  = OFF_Y  + SZ_Y;                       // f16 [B][3136][64]
constexpr size_t SZ_Q   = (size_t)Bsz*Nn*64*2;
constexpr size_t OFF_KT = OFF_Q  + SZ_Q;                       // f16 [B][2][208][32]
constexpr size_t SZ_KT  = (size_t)Bsz*HEADS*NKP*HD*2;
constexpr size_t OFF_VT = OFF_KT + SZ_KT;                      // f16 [B][2][32][224]
constexpr size_t SZ_VT  = (size_t)Bsz*HEADS*HD*NKP2*2;
constexpr size_t OFF_FT = OFF_VT + SZ_VT;                      // f16 [B][208][64]
constexpr size_t SZ_FT  = (size_t)Bsz*NKP*64*2;
constexpr size_t OFF_WD = OFF_FT + SZ_FT;                      // f32 [B][64][9]
constexpr size_t SZ_WD  = (size_t)Bsz*64*9*4;
constexpr size_t OFF_BD = OFF_WD + SZ_WD;                      // f32 [B][64]

__device__ __forceinline__ float gelu_f(float x) {
    return 0.5f * x * (1.0f + erff(x * 0.70710678118654752f));
}

// Build a v16h fragment from two 8-half (16B) chunks.
__device__ __forceinline__ v16h frag_pair(const _Float16* p0, const _Float16* p1) {
    v8h a = *(const v8h*)p0;
    v8h b = *(const v8h*)p1;
    v16h r;
#pragma unroll
    for (int i = 0; i < 8; ++i) { r[i] = a[i]; r[i + 8] = b[i]; }
    return r;
}

__device__ __forceinline__ v8f wmma_f16(v16h a, v16h b, v8f c) {
    return __builtin_amdgcn_wmma_f32_16x16x32_f16(false, a, false, b, (short)0, c, false, false);
}

// CDNA5 async global->LDS copy (16B per lane). Tracked by ASYNCcnt.
__device__ __forceinline__ void async_copy_b128(unsigned lds_off, const void* gptr) {
    asm volatile("global_load_async_to_lds_b128 %0, %1, off"
                 :: "v"(lds_off), "v"(gptr) : "memory");
}
__device__ __forceinline__ void wait_asynccnt0() {
    asm volatile("s_wait_asynccnt 0x0" ::: "memory");
}

// ============================================================
// 1) Dynamic-conv weight generation: pool + MLP + softmax mix
// ============================================================
__global__ void k_dynw(const float* __restrict__ x,
                       const float* __restrict__ dc_w, const float* __restrict__ dc_b,
                       const float* __restrict__ dcp1_w, const float* __restrict__ dcp1_g,
                       const float* __restrict__ dcp1_b,
                       const float* __restrict__ dcp2_w, const float* __restrict__ dcp2_b,
                       float* __restrict__ wdyn, float* __restrict__ bdyn) {
    __shared__ float s_pool[64 * 10];   // token 0..8 = 3x3 bins, 9 = global mean
    __shared__ float s_hid[16 * 10];
    __shared__ float s_out[128 * 10];
    const int b = blockIdx.x, tid = threadIdx.x;
    const float* x1 = x + (size_t)b * Cc * Nn;  // channels 0..63

    for (int idx = tid; idx < 640; idx += 256) {
        float acc = 0.f;
        if (idx < 576) {
            int c = idx / 9, bin = idx % 9, i = bin / 3, j = bin % 3;
            int hs = (i * Hh) / 3, he = ((i + 1) * Hh + 2) / 3;
            int cs = (j * Ww) / 3, ce = ((j + 1) * Ww + 2) / 3;
            const float* p = x1 + (size_t)c * Nn;
            for (int r = hs; r < he; ++r)
                for (int q = cs; q < ce; ++q) acc += p[r * Ww + q];
            s_pool[c * 10 + bin] = acc / (float)((he - hs) * (ce - cs));
        } else {
            int c = idx - 576;
            const float* p = x1 + (size_t)c * Nn;
            for (int r = 0; r < Nn; ++r) acc += p[r];
            s_pool[c * 10 + 9] = acc / (float)Nn;
        }
    }
    __syncthreads();
    for (int idx = tid; idx < 160; idx += 256) {
        int h = idx / 10, tok = idx % 10;
        float acc = 0.f;
        for (int c = 0; c < 64; ++c) acc += dcp1_w[h * 64 + c] * s_pool[c * 10 + tok];
        s_hid[h * 10 + tok] = gelu_f(acc * dcp1_g[h] + dcp1_b[h]);
    }
    __syncthreads();
    for (int idx = tid; idx < 1280; idx += 256) {
        int o = idx / 10, tok = idx % 10;
        float acc = dcp2_b[o];
        for (int h = 0; h < 16; ++h) acc += dcp2_w[o * 16 + h] * s_hid[h * 10 + tok];
        s_out[o * 10 + tok] = acc;
    }
    __syncthreads();
    for (int idx = tid; idx < 640; idx += 256) {
        if (idx < 576) {
            int c = idx / 9, bin = idx % 9;
            float a0 = s_out[c * 10 + bin], a1 = s_out[(64 + c) * 10 + bin];
            float m = fmaxf(a0, a1);
            float e0 = expf(a0 - m), e1 = expf(a1 - m), inv = 1.f / (e0 + e1);
            wdyn[((size_t)b * 64 + c) * 9 + bin] =
                (e0 * inv) * dc_w[c * 9 + bin] + (e1 * inv) * dc_w[576 + c * 9 + bin];
        } else {
            int c = idx - 576;
            float a0 = s_out[c * 10 + 9], a1 = s_out[(64 + c) * 10 + 9];
            float m = fmaxf(a0, a1);
            float e0 = expf(a0 - m), e1 = expf(a1 - m), inv = 1.f / (e0 + e1);
            bdyn[b * 64 + c] = (e0 * inv) * dc_b[c] + (e1 * inv) * dc_b[64 + c];
        }
    }
}

// ============================================================
// 2) Dynamic depthwise 3x3 conv -> y channels [0..63]
// ============================================================
__global__ void k_y1(const float* __restrict__ x, const float* __restrict__ wdyn,
                     const float* __restrict__ bdyn, float* __restrict__ y) {
    const int bc = blockIdx.x, b = bc >> 6, c = bc & 63;
    const float* xin = x + ((size_t)b * Cc + c) * Nn;
    float w[9];
#pragma unroll
    for (int k = 0; k < 9; ++k) w[k] = wdyn[(size_t)bc * 9 + k];
    const float bias = bdyn[bc];
    float* yo = y + ((size_t)b * Cc + c) * Nn;
    for (int p = threadIdx.x; p < Nn; p += 256) {
        int i = p / Ww, j = p % Ww;
        float acc = bias;
#pragma unroll
        for (int di = 0; di < 3; ++di) {
            int ii = i + di - 1;
            if (ii < 0 || ii >= Hh) continue;
#pragma unroll
            for (int dj = 0; dj < 3; ++dj) {
                int jj = j + dj - 1;
                if (jj < 0 || jj >= Ww) continue;
                acc += xin[ii * Ww + jj] * w[di * 3 + dj];
            }
        }
        yo[p] = acc;
    }
}

// ============================================================
// 3) kv spatial-reduction path -> featT f16 [b][208][64]
// ============================================================
__global__ void k_feat(const float* __restrict__ x,
                       const float* __restrict__ sr1_w, const float* __restrict__ sr1_g,
                       const float* __restrict__ sr1_b, const float* __restrict__ sr2_w,
                       const float* __restrict__ sr2_g, const float* __restrict__ sr2_b,
                       const float* __restrict__ lc_w, const float* __restrict__ lc_b,
                       _Float16* __restrict__ featT) {
    __shared__ float s_u[64 * NK];
    const int b = blockIdx.x, tid = threadIdx.x;
    const float* x2 = x + ((size_t)b * Cc + CH) * Nn;
    for (int idx = tid; idx < 64 * NK; idx += 256) {
        int c = idx / NK, p = idx % NK, oi = p / 14, oj = p % 14;
        const float* xc = x2 + (size_t)c * Nn;
        const float* wc = sr1_w + c * 49;
        float acc = 0.f;
        for (int r = 0; r < 7; ++r) {
            int ii = oi * 4 - 3 + r;
            if (ii < 0 || ii >= Hh) continue;
            for (int s = 0; s < 7; ++s) {
                int jj = oj * 4 - 3 + s;
                if (jj < 0 || jj >= Ww) continue;
                acc += xc[ii * Ww + jj] * wc[r * 7 + s];
            }
        }
        float v = gelu_f(acc * sr1_g[c] + sr1_b[c]);
        v = v * sr2_w[c] * sr2_g[c] + sr2_b[c];
        s_u[idx] = v;
    }
    __syncthreads();
    _Float16* ft = featT + (size_t)b * NKP * 64;
    for (int idx = tid; idx < 64 * NK; idx += 256) {
        int c = idx / NK, p = idx % NK, oi = p / 14, oj = p % 14;
        const float* uc = s_u + c * NK;
        const float* wc = lc_w + c * 9;
        float acc = lc_b[c] + uc[p];
#pragma unroll
        for (int di = 0; di < 3; ++di) {
            int ii = oi + di - 1;
            if (ii < 0 || ii >= 14) continue;
#pragma unroll
            for (int dj = 0; dj < 3; ++dj) {
                int jj = oj + dj - 1;
                if (jj < 0 || jj >= 14) continue;
                acc += uc[ii * 14 + jj] * wc[di * 3 + dj];
            }
        }
        ft[p * 64 + c] = (_Float16)acc;
    }
    for (int idx = tid; idx < 12 * 64; idx += 256) {   // zero pad tokens 196..207
        ft[(NK + idx / 64) * 64 + (idx % 64)] = (_Float16)0.f;
    }
}

// ============================================================
// 4) kv pointwise projection (WMMA): kT / vT layouts for attention
//    waves 0-3 -> k half (packed stores), waves 4-7 -> v half
// ============================================================
__global__ void k_kvproj(const float* __restrict__ kv_w, const float* __restrict__ kv_b,
                         const _Float16* __restrict__ featT,
                         _Float16* __restrict__ kT, _Float16* __restrict__ vT) {
    __shared__ __align__(16) _Float16 s_w[128 * 64];
    const int b = blockIdx.x, tid = threadIdx.x;
    for (int idx = tid; idx < 128 * 64; idx += 256) s_w[idx] = (_Float16)kv_w[idx];
    __syncthreads();
    const int wave = tid >> 5, lane = tid & 31, r = lane & 15, hi = (lane >> 4) & 1;
    const int m0 = wave * 16;
    const _Float16* ft = featT + (size_t)b * NKP * 64;
    _Float16* kTb = kT + (size_t)b * HEADS * NKP * HD;
    _Float16* vTb = vT + (size_t)b * HEADS * HD * NKP2;
    const bool is_k = (m0 < 64);          // wave-uniform split
    for (int jt = 0; jt < 13; ++jt) {
        const int n0 = jt * 16;
        v8f acc = {0.f, 0.f, 0.f, 0.f, 0.f, 0.f, 0.f, 0.f};
#pragma unroll
        for (int ks = 0; ks < 2; ++ks) {
            const int k0 = ks * 32;
            const _Float16* ap = s_w + (m0 + r) * 64 + k0 + (hi ? 8 : 0);   // A: out x chan
            v16h a = frag_pair(ap, ap + 16);
            const _Float16* bp = ft + (n0 + r) * 64 + k0 + (hi ? 8 : 0);   // B: chan x tok
            v16h bf = frag_pair(bp, bp + 16);
            acc = wmma_f16(a, bf, acc);
        }
        const int t = n0 + r;
        const int mbase = m0 + (hi ? 8 : 0);
        if (is_k) {
            if (t < NK) {
                const int hh = mbase >> 5, dbase = mbase & 31;
                v8h pk;
#pragma unroll
                for (int rr = 0; rr < 8; ++rr)
                    pk[rr] = (_Float16)(acc[rr] + kv_b[mbase + rr]);
                *(v8h*)(kTb + ((size_t)hh * NKP + t) * HD + dbase) = pk;   // 16B packed
            }
        } else {
            if (t < NK) {
                const int o2 = mbase - 64, hh = o2 >> 5, dbase = o2 & 31;
#pragma unroll
                for (int rr = 0; rr < 8; ++rr)
                    vTb[((size_t)hh * HD + dbase + rr) * NKP2 + t] =
                        (_Float16)(acc[rr] + kv_b[mbase + rr]);
            }
        }
    }
    for (int idx = tid; idx < HEADS * 12 * HD; idx += 256) {   // kT pad tokens
        int hh = idx / (12 * HD), rem = idx % (12 * HD);
        kTb[(hh * NKP + NK + rem / HD) * HD + (rem % HD)] = (_Float16)0.f;
    }
    for (int idx = tid; idx < HEADS * HD * 28; idx += 256) {   // vT pad tokens
        int hh = idx / (HD * 28), rem = idx % (HD * 28);
        vTb[(hh * HD + rem / 28) * NKP2 + NK + (rem % 28)] = (_Float16)0.f;
    }
}

// ============================================================
// 5) q pointwise projection (WMMA) -> qf16 [b][n][64]
// ============================================================
__global__ void k_qproj(const float* __restrict__ x, const float* __restrict__ q_w,
                        const float* __restrict__ q_b, _Float16* __restrict__ qf) {
    __shared__ __align__(16) _Float16 s_wq[64 * 64];
    __shared__ __align__(16) _Float16 s_x[64 * 64];   // [token][chan]
    const int blk = blockIdx.x, b = blk / 49, nbase = (blk % 49) * 64;
    const int tid = threadIdx.x;
    for (int idx = tid; idx < 4096; idx += 128) s_wq[idx] = (_Float16)q_w[idx];
    const float* x2 = x + ((size_t)b * Cc + CH) * Nn + nbase;
    for (int idx = tid; idx < 4096; idx += 128) {
        int c = idx >> 6, t = idx & 63;
        s_x[t * 64 + c] = (_Float16)x2[(size_t)c * Nn + t];
    }
    __syncthreads();
    const int wave = tid >> 5, lane = tid & 31, r = lane & 15, hi = (lane >> 4) & 1;
    const int m0 = wave * 16;
    _Float16* qb = qf + ((size_t)b * Nn + nbase) * 64;
    for (int ot = 0; ot < 4; ++ot) {
        const int o0 = ot * 16;
        v8f acc = {0.f, 0.f, 0.f, 0.f, 0.f, 0.f, 0.f, 0.f};
#pragma unroll
        for (int ks = 0; ks < 2; ++ks) {
            const int k0 = ks * 32;
            const _Float16* ap = s_x + (m0 + r) * 64 + k0 + (hi ? 8 : 0);   // A: tok x chan
            v16h a = frag_pair(ap, ap + 16);
            const _Float16* bp = s_wq + (o0 + r) * 64 + k0 + (hi ? 8 : 0); // B: chan x out
            v16h bf = frag_pair(bp, bp + 16);
            acc = wmma_f16(a, bf, acc);
        }
        const int o = o0 + r;
        const float bias = q_b[o];
        const int mbase = m0 + (hi ? 8 : 0);
#pragma unroll
        for (int rr = 0; rr < 8; ++rr)
            qb[(size_t)(mbase + rr) * 64 + o] = (_Float16)(acc[rr] + bias);
    }
}

// ============================================================
// 6) Attention core: one wave per 16-row q tile (WMMA + softmax)
//    rpe block prefetched into LDS via async copy, overlapping q@kT
// ============================================================
__global__ void k_attn(const _Float16* __restrict__ qf, const _Float16* __restrict__ kT,
                       const _Float16* __restrict__ vT, const float* __restrict__ rpe,
                       float* __restrict__ y) {
    __shared__ float s_S[16 * NKP];                    // S scores f32
    __shared__ __align__(16) _Float16 s_P[16 * NKP2];  // probabilities f16
    __shared__ __align__(16) float s_rpe[16 * NK];     // rpe tile (async-filled)
    const int blk = blockIdx.x;
    const int qt = blk % 196, bh = blk / 196, b = bh >> 1, h = bh & 1;
    const int n0 = qt * 16;
    const int lane = threadIdx.x, r = lane & 15, hi = (lane >> 4) & 1;

    // Kick off async copy of the contiguous 16x196 f32 rpe block into LDS.
    // 784 x 16B transfers; overlaps with the q@kT WMMA phase below.
    {
        const char* src = (const char*)(rpe + ((size_t)h * Nn + n0) * NK);
        unsigned dst = (unsigned)(size_t)(void*)&s_rpe[0];
        for (int idx = lane; idx < 784; idx += 32)
            async_copy_b128(dst + (unsigned)idx * 16u, src + (size_t)idx * 16);
    }

    const _Float16* vTb = vT + (size_t)(b * HEADS + h) * HD * NKP2;
    __builtin_prefetch(vTb + (size_t)lane * NKP2, 0, 0);   // warm vT panel

    // A fragment: q tile 16x32 (reused across all 13 column tiles)
    const _Float16* qp = qf + ((size_t)b * Nn + n0 + r) * 64 + h * HD + (hi ? 8 : 0);
    const v16h aq = frag_pair(qp, qp + 16);
    const _Float16* kTb = kT + (size_t)(b * HEADS + h) * NKP * HD;

    for (int jt = 0; jt < 13; ++jt) {
        const int c0 = jt * 16;
        const _Float16* bp = kTb + (size_t)(c0 + r) * HD + (hi ? 8 : 0);
        v16h bf = frag_pair(bp, bp + 16);
        v8f acc = {0.f, 0.f, 0.f, 0.f, 0.f, 0.f, 0.f, 0.f};
        acc = wmma_f16(aq, bf, acc);
        const int col = c0 + r, mbase = hi ? 8 : 0;
#pragma unroll
        for (int rr = 0; rr < 8; ++rr) s_S[(mbase + rr) * NKP + col] = acc[rr];
    }
    wait_asynccnt0();          // rpe tile resident in LDS
    __syncthreads();

    const float sc = 0.17677669529663688f;   // 32^-0.5
    if (lane < 16) {
        const int row = lane;
        const float* rp = s_rpe + row * NK;
        float* srow = s_S + row * NKP;
        float m = -3.0e38f;
        for (int c = 0; c < NK; ++c) {
            float v = srow[c] * sc + rp[c];
            srow[c] = v;
            m = fmaxf(m, v);
        }
        float sum = 0.f;
        for (int c = 0; c < NK; ++c) {
            float e = expf(srow[c] - m);
            srow[c] = e;
            sum += e;
        }
        const float inv = 1.f / sum;
        _Float16* prow = s_P + row * NKP2;
        for (int c = 0; c < NK; ++c) prow[c] = (_Float16)(srow[c] * inv);
        for (int c = NK; c < NKP2; ++c) prow[c] = (_Float16)0.f;   // mask pads
    }
    __syncthreads();

    float* yb = y + (size_t)b * Cc * Nn;
    for (int nt = 0; nt < 2; ++nt) {
        const int d0 = nt * 16;
        v8f acc = {0.f, 0.f, 0.f, 0.f, 0.f, 0.f, 0.f, 0.f};
        for (int ks = 0; ks < 7; ++ks) {
            const int k0 = ks * 32;
            const _Float16* ap = s_P + r * NKP2 + k0 + (hi ? 8 : 0);      // A: tok x kTok
            v16h a = frag_pair(ap, ap + 16);
            const _Float16* bp = vTb + (size_t)(d0 + r) * NKP2 + k0 + (hi ? 8 : 0); // B
            v16h bf = frag_pair(bp, bp + 16);
            acc = wmma_f16(a, bf, acc);
        }
        const int ch = CH + h * HD + d0 + r;
        float* yo = yb + (size_t)ch * Nn + n0 + (hi ? 8 : 0);
        f4v lo = {acc[0], acc[1], acc[2], acc[3]};
        f4v hi4 = {acc[4], acc[5], acc[6], acc[7]};
        *(f4v*)yo = lo;
        *(f4v*)(yo + 4) = hi4;
    }
}

// ============================================================
// 7) Fused tail: p1 (dw3x3)+gelu+bn -> p2+gelu+bn -> p3+bn -> +y
// ============================================================
__global__ void k_tail(const float* __restrict__ y,
                       const float* __restrict__ p1_w, const float* __restrict__ p1_b,
                       const float* __restrict__ p1_g, const float* __restrict__ p1_bb,
                       const float* __restrict__ p2_w, const float* __restrict__ p2_b,
                       const float* __restrict__ p2_g, const float* __restrict__ p2_bb,
                       const float* __restrict__ p3_w, const float* __restrict__ p3_b,
                       const float* __restrict__ p3_g, const float* __restrict__ p3_bb,
                       float* __restrict__ out) {
    __shared__ float s_ch[16 * 16 * 16];   // 16 chans of a 16x16 halo tile
    __shared__ float s_w2[16 * 128];
    __shared__ float s_w3[128 * 16];
    const int blk = blockIdx.x, b = blk >> 4, t = blk & 15;
    const int r0 = (t >> 2) * 14, c0 = (t & 3) * 14;
    const int tid = threadIdx.x;
    for (int idx = tid; idx < 2048; idx += 256) { s_w2[idx] = p2_w[idx]; s_w3[idx] = p3_w[idx]; }
    const bool act = tid < 196;
    const int li = tid / 14, lj = tid % 14;
    float acc[16];
#pragma unroll
    for (int i = 0; i < 16; ++i) acc[i] = 0.f;
    const float* yb = y + (size_t)b * Cc * Nn;

    for (int cc = 0; cc < 8; ++cc) {
        __syncthreads();
        const int cbase = cc * 16;
        for (int idx = tid; idx < 4096; idx += 256) {
            int ch = idx >> 8, rem = idx & 255, ii = rem >> 4, jj = rem & 15;
            int gi = r0 - 1 + ii, gj = c0 - 1 + jj;
            float v = 0.f;
            if (gi >= 0 && gi < Hh && gj >= 0 && gj < Ww)
                v = yb[(size_t)(cbase + ch) * Nn + gi * Ww + gj];
            s_ch[idx] = v;
        }
        __syncthreads();
        if (act) {
#pragma unroll
            for (int c = 0; c < 16; ++c) {
                const float* w = p1_w + (cbase + c) * 9;
                const float* tile = s_ch + c * 256;
                float v = p1_b[cbase + c];
#pragma unroll
                for (int di = 0; di < 3; ++di)
#pragma unroll
                    for (int dj = 0; dj < 3; ++dj)
                        v += tile[(li + di) * 16 + (lj + dj)] * w[di * 3 + dj];
                v = gelu_f(v);
                v = v * p1_g[cbase + c] + p1_bb[cbase + c];
#pragma unroll
                for (int hh = 0; hh < 16; ++hh) acc[hh] += s_w2[hh * 128 + cbase + c] * v;
            }
        }
    }
    if (act) {
        float hid[16];
#pragma unroll
        for (int hh = 0; hh < 16; ++hh) {
            float v = gelu_f(acc[hh] + p2_b[hh]);
            hid[hh] = v * p2_g[hh] + p2_bb[hh];
        }
        const int n = (r0 + li) * Ww + (c0 + lj);
        for (int o = 0; o < Cc; ++o) {
            float z = p3_b[o];
#pragma unroll
            for (int hh = 0; hh < 16; ++hh) z += s_w3[o * 16 + hh] * hid[hh];
            z = z * p3_g[o] + p3_bb[o];
            out[((size_t)b * Cc + o) * Nn + n] = z + yb[(size_t)o * Nn + n];
        }
    }
}

// ============================================================
extern "C" void kernel_launch(void* const* d_in, const int* in_sizes, int n_in,
                              void* d_out, int out_size, void* d_ws, size_t ws_size,
                              hipStream_t stream) {
    (void)in_sizes; (void)n_in; (void)out_size; (void)ws_size;
    const float* x       = (const float*)d_in[0];
    const float* rpe     = (const float*)d_in[1];
    const float* dc_w    = (const float*)d_in[2];
    const float* dc_b    = (const float*)d_in[3];
    const float* dcp1_w  = (const float*)d_in[4];
    const float* dcp1_g  = (const float*)d_in[5];
    const float* dcp1_b  = (const float*)d_in[6];
    const float* dcp2_w  = (const float*)d_in[7];
    const float* dcp2_b  = (const float*)d_in[8];
    const float* q_w     = (const float*)d_in[9];
    const float* q_b     = (const float*)d_in[10];
    const float* kv_w    = (const float*)d_in[11];
    const float* kv_b    = (const float*)d_in[12];
    const float* sr1_w   = (const float*)d_in[13];
    const float* sr1_g   = (const float*)d_in[14];
    const float* sr1_b   = (const float*)d_in[15];
    const float* sr2_w   = (const float*)d_in[16];
    const float* sr2_g   = (const float*)d_in[17];
    const float* sr2_b   = (const float*)d_in[18];
    const float* lc_w    = (const float*)d_in[19];
    const float* lc_b    = (const float*)d_in[20];
    const float* p1_w    = (const float*)d_in[21];
    const float* p1_b    = (const float*)d_in[22];
    const float* p1_g    = (const float*)d_in[23];
    const float* p1_bb   = (const float*)d_in[24];
    const float* p2_w    = (const float*)d_in[25];
    const float* p2_b    = (const float*)d_in[26];
    const float* p2_g    = (const float*)d_in[27];
    const float* p2_bb   = (const float*)d_in[28];
    const float* p3_w    = (const float*)d_in[29];
    const float* p3_b    = (const float*)d_in[30];
    const float* p3_g    = (const float*)d_in[31];
    const float* p3_bb   = (const float*)d_in[32];

    char* ws = (char*)d_ws;
    float*    y_ws  = (float*)(ws + OFF_Y);
    _Float16* qf    = (_Float16*)(ws + OFF_Q);
    _Float16* kT    = (_Float16*)(ws + OFF_KT);
    _Float16* vT    = (_Float16*)(ws + OFF_VT);
    _Float16* featT = (_Float16*)(ws + OFF_FT);
    float*    wdyn  = (float*)(ws + OFF_WD);
    float*    bdyn  = (float*)(ws + OFF_BD);
    float*    out   = (float*)d_out;

    k_dynw<<<Bsz, 256, 0, stream>>>(x, dc_w, dc_b, dcp1_w, dcp1_g, dcp1_b,
                                    dcp2_w, dcp2_b, wdyn, bdyn);
    k_y1<<<Bsz * CH, 256, 0, stream>>>(x, wdyn, bdyn, y_ws);
    k_feat<<<Bsz, 256, 0, stream>>>(x, sr1_w, sr1_g, sr1_b, sr2_w, sr2_g, sr2_b,
                                    lc_w, lc_b, featT);
    k_kvproj<<<Bsz, 256, 0, stream>>>(kv_w, kv_b, featT, kT, vT);
    k_qproj<<<Bsz * 49, 128, 0, stream>>>(x, q_w, q_b, qf);
    k_attn<<<Bsz * HEADS * 196, 32, 0, stream>>>(qf, kT, vT, rpe, y_ws);
    k_tail<<<Bsz * 16, 256, 0, stream>>>(y_ws, p1_w, p1_b, p1_g, p1_bb,
                                         p2_w, p2_b, p2_g, p2_bb,
                                         p3_w, p3_b, p3_g, p3_bb, out);
}
